// SPECTEncoder_46084999086400
// MI455X (gfx1250) — matrix operand order repaired and checked by reading, not compiled, not run on hardware
//
#include <hip/hip_runtime.h>
#include <hip/hip_bf16.h>

// ---------------------------------------------------------------------------
// SPECT GNN encoder, fully dense-rewritten for MI455X (gfx1250, wave32, WMMA).
// Complete-graph GCN => per-graph mean => three dense GEMMs + BN + L2 norm.
// GEMMs run on v_wmma_f32_16x16x32_bf16 (f32 accumulate); A tiles staged to
// LDS via global_load_async_to_lds_b128 (ASYNCcnt path).
// ---------------------------------------------------------------------------

typedef __bf16 bf16_t;
typedef bf16_t v16bf __attribute__((ext_vector_type(16)));
typedef bf16_t v8bf  __attribute__((ext_vector_type(8)));
typedef float  v8f   __attribute__((ext_vector_type(8)));

union V16 { v16bf v; v8bf h[2]; };

#define NG   32768   // graphs
#define HID  128
#define P1   512
#define EMB  1024

__device__ __forceinline__ float lrelu(float v) { return fmaxf(v, 0.2f * v); }

__device__ __forceinline__ v8f wmma_bf16(const V16& a, const V16& b, v8f c) {
  return __builtin_amdgcn_wmma_f32_16x16x32_bf16(
      /*neg_a=*/false, a.v, /*neg_b=*/false, b.v,
      /*c_mod=*/(short)0, c, /*reuse_a=*/false, /*reuse_b=*/false);
}

// A tile: row-major bf16 [rows][stride] in LDS.
// Lane holds row=lane&15; lo lanes K {base..base+7, base+16..23}, hi lanes +8.
__device__ __forceinline__ V16 loadA(const bf16_t* s, int row, int stride,
                                     int kbase, int hi) {
  V16 a;
  const bf16_t* p = s + row * stride + kbase + (hi ? 8 : 0);
  a.h[0] = *(const v8bf*)(p);
  a.h[1] = *(const v8bf*)(p + 16);
  return a;
}

// B tile: weights stored [N][K] bf16 row-major in global.
// Lane holds col=lane&15; lo lanes K {base..base+7, base+16..23}, hi lanes +8.
__device__ __forceinline__ V16 loadB(const bf16_t* w, int col, int K,
                                     int kbase, int hi) {
  V16 b;
  const bf16_t* p = w + (size_t)col * K + kbase + (hi ? 8 : 0);
  b.h[0] = *(const v8bf*)(p);
  b.h[1] = *(const v8bf*)(p + 16);
  return b;
}

// Async copy of one 16-byte chunk global -> LDS (per-lane), ASYNCcnt tracked.
__device__ __forceinline__ void async_b128(unsigned lds_off, const void* gaddr) {
  asm volatile("global_load_async_to_lds_b128 %0, %1, off"
               :: "v"(lds_off), "v"((unsigned long long)gaddr) : "memory");
}
__device__ __forceinline__ void wait_async0() {
  asm volatile("s_wait_asynccnt 0x0" ::: "memory");
}

// --- weight convert: src f32 [K][N] row-major -> dst bf16 [N][K] -------------
__global__ void kConvT(const float* __restrict__ src, bf16_t* __restrict__ dst,
                       int K, int N) {
  int tid = blockIdx.x * blockDim.x + threadIdx.x;
  if (tid >= K * N) return;
  int k = tid % K, n = tid / K;
  dst[tid] = (bf16_t)src[(size_t)k * N + n];
}

// --- stage A: per-graph mean -> rank-1 conv1 -> WMMA conv2 -> z (bf16) -------
__global__ void kA(const float* __restrict__ x, const float* __restrict__ roi,
                   const float* __restrict__ W1, const float* __restrict__ b1,
                   const bf16_t* __restrict__ W2b, const float* __restrict__ b2,
                   bf16_t* __restrict__ zb) {
  __shared__ __align__(16) bf16_t sA[16 * HID];
  __shared__ float sM[16];
  int t = threadIdx.x;
  int g0 = blockIdx.x * 16;
  if (t < 16) {                       // per-graph mean of x * roi_scaler
    float s = 0.f;
    int base = (g0 + t) * 7;
    #pragma unroll
    for (int k = 0; k < 7; k++) s += x[base + k] * roi[k];
    sM[t] = s * (1.0f / 7.0f);
  }
  __syncthreads();
  {                                   // h1 = lrelu(m * W1 + b1), bf16 tile
    int r = t >> 4, c0 = (t & 15) * 8;
    float m = sM[r];
    #pragma unroll
    for (int j = 0; j < 8; j++) {
      int c = c0 + j;
      sA[r * HID + c] = (bf16_t)lrelu(m * W1[c] + b1[c]);
    }
  }
  __syncthreads();
  int lane = t & 31, w = t >> 5, hi = lane >> 4, rc = lane & 15;
  int n0 = w * 16;                    // 8 waves cover 128 output cols
  v8f acc = {};
  #pragma unroll
  for (int kk = 0; kk < 4; kk++) {
    V16 a = loadA(sA, rc, HID, kk * 32, hi);
    V16 b = loadB(W2b, n0 + rc, HID, kk * 32, hi);
    acc = wmma_bf16(a, b, acc);
  }
  float bias = b2[n0 + rc];
  #pragma unroll
  for (int v = 0; v < 8; v++) {
    int mr = v + hi * 8;
    zb[(size_t)(g0 + mr) * HID + n0 + rc] = (bf16_t)lrelu(acc[v] + bias);
  }
}

// --- stage B: p_raw = z @ Wp1 + bp1 (bf16 out, pre-BN) -----------------------
__global__ void kB(const bf16_t* __restrict__ zb, const bf16_t* __restrict__ Wp1b,
                   const float* __restrict__ bp1, bf16_t* __restrict__ p) {
  __shared__ __align__(16) bf16_t sA[16 * HID];
  int t = threadIdx.x;
  int g0 = blockIdx.x * 16;
  {                                   // async stage 16x128 bf16 (4 KB, contig)
    unsigned ldsbase = (unsigned)(uintptr_t)sA;
    async_b128(ldsbase + (unsigned)t * 16u, zb + (size_t)g0 * HID + t * 8);
    wait_async0();
  }
  __syncthreads();
  int lane = t & 31, w = t >> 5, hi = lane >> 4, rc = lane & 15;
  int n0 = blockIdx.y * 128 + w * 16;
  v8f acc = {};
  #pragma unroll
  for (int kk = 0; kk < 4; kk++) {
    V16 a = loadA(sA, rc, HID, kk * 32, hi);
    V16 b = loadB(Wp1b, n0 + rc, HID, kk * 32, hi);
    acc = wmma_bf16(a, b, acc);
  }
  float bias = bp1[n0 + rc];
  #pragma unroll
  for (int v = 0; v < 8; v++) {
    int mr = v + hi * 8;
    p[(size_t)(g0 + mr) * P1 + n0 + rc] = (bf16_t)(acc[v] + bias);
  }
}

// --- BN batch statistics: column sum / sumsq over 32768 rows -----------------
__global__ void kStats(const bf16_t* __restrict__ p, float* __restrict__ colsum,
                       float* __restrict__ colsq) {
  int c = threadIdx.x;                // 512 threads = 512 columns
  int r0 = blockIdx.x * 64;           // 512 blocks x 64 rows
  float s = 0.f, q = 0.f;
  for (int r = 0; r < 64; r++) {
    float v = (float)p[(size_t)(r0 + r) * P1 + c];
    s += v; q += v * v;
  }
  atomicAdd(&colsum[c], s);
  atomicAdd(&colsq[c], q);
}

__global__ void kBnFin(const float* __restrict__ colsum, const float* __restrict__ colsq,
                       const float* __restrict__ gamma, const float* __restrict__ beta,
                       float* __restrict__ scale, float* __restrict__ shift) {
  int c = threadIdx.x;
  float mu   = colsum[c] * (1.0f / NG);
  float var  = colsq[c] * (1.0f / NG) - mu * mu;
  float rstd = rsqrtf(var + 1e-5f);
  float sc   = gamma[c] * rstd;
  scale[c] = sc;
  shift[c] = beta[c] - mu * sc;
}

// --- BN + leaky-relu, in place (elementwise, same address read/write) --------
__global__ void kBn(bf16_t* __restrict__ p, const float* __restrict__ scale,
                    const float* __restrict__ shift) {
  size_t i = ((size_t)blockIdx.x * blockDim.x + threadIdx.x) * 4;
  int c = (int)(i & (P1 - 1));
  #pragma unroll
  for (int j = 0; j < 4; j++) {
    float v = (float)p[i + j];
    p[i + j] = (bf16_t)lrelu(v * scale[c + j] + shift[c + j]);
  }
}

// --- stage C: out = p @ Wp2 + bp2, fused row-wise L2 normalize ---------------
// 32 graphs per block: each B fragment feeds two WMMAs (halves L2 weight
// traffic vs a 16-row tile).
__global__ void kC(const bf16_t* __restrict__ p, const bf16_t* __restrict__ Wp2b,
                   const float* __restrict__ bp2, float* __restrict__ out) {
  __shared__ __align__(16) bf16_t sA[32 * P1];   // 32 KB
  __shared__ float rowsq[32];
  int t = threadIdx.x;
  int g0 = blockIdx.x * 32;
  if (t < 32) rowsq[t] = 0.f;
  {                                   // async stage 32x512 bf16 (32 KB, contig)
    unsigned ldsbase = (unsigned)(uintptr_t)sA;
    const bf16_t* gbase = p + (size_t)g0 * P1;
    #pragma unroll
    for (int j = 0; j < 8; j++) {
      int e = t + j * 256;
      async_b128(ldsbase + (unsigned)e * 16u, gbase + e * 8);
    }
    wait_async0();
  }
  __syncthreads();
  int lane = t & 31, w = t >> 5, hi = lane >> 4, rc = lane & 15;
  v8f accs0[8], accs1[8];
  float part0[8], part1[8];
  #pragma unroll
  for (int v = 0; v < 8; v++) { part0[v] = 0.f; part1[v] = 0.f; }
  #pragma unroll
  for (int nt = 0; nt < 8; nt++) {    // 8 waves x 8 tiles = 1024 cols
    int n0 = w * 16 + nt * 128;
    v8f a0 = {}, a1 = {};
    #pragma unroll
    for (int kk = 0; kk < 16; kk++) { // K = 512
      V16 b  = loadB(Wp2b, n0 + rc, P1, kk * 32, hi);
      V16 x0 = loadA(sA, rc, P1, kk * 32, hi);            // rows 0..15
      V16 x1 = loadA(sA + 16 * P1, rc, P1, kk * 32, hi);  // rows 16..31
      a0 = wmma_bf16(x0, b, a0);
      a1 = wmma_bf16(x1, b, a1);
    }
    float bias = bp2[n0 + rc];
    #pragma unroll
    for (int v = 0; v < 8; v++) {
      float v0 = a0[v] + bias;
      float v1 = a1[v] + bias;
      a0[v] = v0; a1[v] = v1;
      part0[v] += v0 * v0;
      part1[v] += v1 * v1;
    }
    accs0[nt] = a0;
    accs1[nt] = a1;
  }
  #pragma unroll
  for (int v = 0; v < 8; v++) {
    atomicAdd(&rowsq[v + hi * 8], part0[v]);
    atomicAdd(&rowsq[16 + v + hi * 8], part1[v]);
  }
  __syncthreads();
  #pragma unroll
  for (int v = 0; v < 8; v++) {
    int mr0 = v + hi * 8;
    int mr1 = 16 + mr0;
    float s0 = 1.0f / fmaxf(sqrtf(rowsq[mr0]), 1e-12f);
    float s1 = 1.0f / fmaxf(sqrtf(rowsq[mr1]), 1e-12f);
    #pragma unroll
    for (int nt = 0; nt < 8; nt++) {
      int n0 = w * 16 + nt * 128;
      out[(size_t)(g0 + mr0) * EMB + n0 + rc] = accs0[nt][v] * s0;
      out[(size_t)(g0 + mr1) * EMB + n0 + rc] = accs1[nt][v] * s1;
    }
  }
}

extern "C" void kernel_launch(void* const* d_in, const int* in_sizes, int n_in,
                              void* d_out, int out_size, void* d_ws, size_t ws_size,
                              hipStream_t stream) {
  (void)in_sizes; (void)n_in; (void)out_size; (void)ws_size;
  const float* x     = (const float*)d_in[0];
  // d_in[1] edge_index, d_in[2] batch: topology is fixed -> not needed
  const float* roi   = (const float*)d_in[3];
  const float* W1    = (const float*)d_in[4];
  const float* b1    = (const float*)d_in[5];
  const float* W2    = (const float*)d_in[6];
  const float* b2    = (const float*)d_in[7];
  const float* Wp1   = (const float*)d_in[8];
  const float* bp1   = (const float*)d_in[9];
  const float* gamma = (const float*)d_in[10];
  const float* beta  = (const float*)d_in[11];
  const float* Wp2   = (const float*)d_in[12];
  const float* bp2   = (const float*)d_in[13];
  float* out = (float*)d_out;

  char* ws = (char*)d_ws;
  bf16_t* W2b     = (bf16_t*)(ws);                     //   32 KB
  bf16_t* Wp1b    = (bf16_t*)(ws + 32768);             //  128 KB
  bf16_t* Wp2b    = (bf16_t*)(ws + 163840);            //    1 MB
  bf16_t* zb      = (bf16_t*)(ws + 1212416);           //    8 MB  [G][128]
  bf16_t* pbuf    = (bf16_t*)(ws + 9601024);           //   32 MB  [G][512]
  float*  colsum  = (float*)(ws + 43155456);           //  2 KB
  float*  colsq   = (float*)(ws + 43157504);           //  2 KB
  float*  bnscale = (float*)(ws + 43159552);           //  2 KB
  float*  bnshift = (float*)(ws + 43161600);           //  2 KB

  hipMemsetAsync(colsum, 0, 4096, stream);             // colsum + colsq

  kConvT<<<(HID * HID + 255) / 256, 256, 0, stream>>>(W2, W2b, HID, HID);
  kConvT<<<(HID * P1 + 255) / 256, 256, 0, stream>>>(Wp1, Wp1b, HID, P1);
  kConvT<<<(P1 * EMB + 255) / 256, 256, 0, stream>>>(Wp2, Wp2b, P1, EMB);

  kA<<<NG / 16, 256, 0, stream>>>(x, roi, W1, b1, W2b, b2, zb);
  kB<<<dim3(NG / 16, P1 / 128), 256, 0, stream>>>(zb, Wp1b, bp1, pbuf);
  kStats<<<NG / 64, 512, 0, stream>>>(pbuf, colsum, colsq);
  kBnFin<<<1, 512, 0, stream>>>(colsum, colsq, gamma, beta, bnscale, bnshift);
  kBn<<<(NG * P1) / 1024, 256, 0, stream>>>(pbuf, bnscale, bnshift);
  kC<<<NG / 32, 256, 0, stream>>>(pbuf, Wp2b, bp2, out);
}
